// finalModel_12317966205008
// MI455X (gfx1250) — compile-verified, matrix operand order checked by reading, not run on hardware
//
#include <hip/hip_runtime.h>
#include <hip/hip_bf16.h>

// ---------------- types ----------------
typedef __bf16 bf16_t;
typedef bf16_t v16bf __attribute__((ext_vector_type(16)));
typedef float  v8f   __attribute__((ext_vector_type(8)));

#if defined(__HIP_DEVICE_COMPILE__) && \
    __has_builtin(__builtin_amdgcn_global_load_async_to_lds_b128) && \
    __has_builtin(__builtin_amdgcn_s_wait_asynccnt)
#define USE_ASYNC_LDS 1
#else
#define USE_ASYNC_LDS 0
#endif

#if USE_ASYNC_LDS
typedef int v4i __attribute__((vector_size(16)));
typedef __attribute__((address_space(1))) v4i gv4i;
typedef __attribute__((address_space(3))) v4i lv4i;
__device__ __forceinline__ gv4i* to_glb(const void* p) { return (gv4i*)p; }
__device__ __forceinline__ lv4i* to_lds(void* p)       { return (lv4i*)p; }
#endif

// ---------------- problem constants ----------------
constexpr int cB   = 4;
constexpr int cT   = 512;
constexpr int cH   = 768;
constexpr int cN   = 256;   // nodes
constexpr int cS   = 16;    // span positions
constexpr int cNERD= 32;
constexpr int cD   = 800;   // H + NER_DIM
constexpr int cL   = 2;
constexpr int cR   = 15;
constexpr int cRS  = 16;    // R + 1 (self loop folded in)
constexpr int cNE  = 64;
constexpr int cMM  = 8;
constexpr int cP   = 1024;
constexpr int cE   = 4096;
constexpr int cOUT = 97;
constexpr int cEE  = 2 * cD;      // 1600
constexpr int cF4  = 4 * cEE;     // 6400
constexpr int cHID = 2 * cEE;     // 3200
constexpr long long cKGNN = (long long)cRS * cD; // 12800

// ---------------- small helper kernels ----------------
__global__ void k_f32_to_bf16(const float* __restrict__ s, bf16_t* __restrict__ d, long long n) {
    long long i = (long long)blockIdx.x * blockDim.x + threadIdx.x;
    long long stride = (long long)gridDim.x * blockDim.x;
    for (; i < n; i += stride) d[i] = (bf16_t)s[i];
}

// W (K,N) f32 -> WT (N,K) bf16, LDS-tiled transpose, both sides coalesced
__global__ void k_transpose_bf16(const float* __restrict__ W, bf16_t* __restrict__ WT,
                                 int K, int N) {
    __shared__ float tile[32][33];
    int k0 = blockIdx.y * 32, n0 = blockIdx.x * 32;
    int lane = threadIdx.x & 31, row = threadIdx.x >> 5;
    for (int j = 0; j < 4; ++j) {
        int r = row + j * 8;
        int k = k0 + r, n = n0 + lane;
        tile[r][lane] = (k < K && n < N) ? W[(long long)k * N + n] : 0.f;
    }
    __syncthreads();
    for (int j = 0; j < 4; ++j) {
        int r = row + j * 8;
        int n = n0 + r, k = k0 + lane;
        if (n < N && k < K) WT[(long long)n * K + k] = (bf16_t)tile[lane][r];
    }
}

// WcatT[l][e][(r*D+d)] : transposed concat of W_rel (r<15) and W_self (r==15)
__global__ void k_build_wcatT(const float* __restrict__ Wrel, const float* __restrict__ Wself,
                              bf16_t* __restrict__ wcatT) {
    __shared__ float tile[32][33];
    int l   = blockIdx.z;
    int kk0 = blockIdx.y * 32;   // K = r*800+d (12800), divisible by 32
    int e0  = blockIdx.x * 32;   // N = 800, divisible by 32
    int lane = threadIdx.x & 31, row = threadIdx.x >> 5;
    for (int j = 0; j < 4; ++j) {
        int rr = row + j * 8;
        int kk = kk0 + rr;
        int r = kk / cD, d = kk % cD;
        const float* src = (r < cR)
            ? &Wrel[(((long long)(l * cR + r)) * cD + d) * cD]
            : &Wself[(((long long)l) * cD + d) * cD];
        tile[rr][lane] = src[e0 + lane];
    }
    __syncthreads();
    for (int j = 0; j < 4; ++j) {
        int rr = row + j * 8;
        long long e = e0 + rr, k = kk0 + lane;
        wcatT[((long long)l * cD + e) * cKGNN + k] = (bf16_t)tile[lane][rr];
    }
}

// doc_proj = relu(doc_cls @ doc_W + doc_b)  -> (B, D)
__global__ void k_doc_proj(const float* __restrict__ doccls, const float* __restrict__ docW,
                           const float* __restrict__ docb, float* __restrict__ docp) {
    int t = blockIdx.x * blockDim.x + threadIdx.x;
    if (t >= cB * cD) return;
    int b = t / cD, d = t % cD;
    float acc = docb[d];
    const float* cl = doccls + (long long)b * cH;
    for (int h = 0; h < cH; ++h) acc += cl[h] * docW[(long long)h * cD + d];
    docp[t] = acc > 0.f ? acc : 0.f;
}

// node feature build -> h0 (B,N,D)
__global__ void k_build_nodes(const float* __restrict__ tokf, const float* __restrict__ doccls,
                              const int* __restrict__ spanpos, const float* __restrict__ spanmask,
                              const int* __restrict__ nner, const float* __restrict__ nmask,
                              const int* __restrict__ mentnum, const float* __restrict__ nerEmb,
                              const float* __restrict__ docp, float* __restrict__ h0) {
    int blk = blockIdx.x;
    int b = blk / cN, n = blk % cN;
    int ment = mentnum[b];
    const int*   pos = spanpos  + ((long long)(b * cN + n)) * cS;
    const float* sm  = spanmask + ((long long)(b * cN + n)) * cS;
    float msum = 0.f;
    for (int s = 0; s < cS; ++s) msum += sm[s];
    float inv = 1.f / (msum + 0.1f);
    float nm  = nmask[b * cN + n];
    int ner   = nner[b * cN + n];
    float* out = h0 + ((long long)(b * cN + n)) * cD;
    for (int d = threadIdx.x; d < cD; d += blockDim.x) {
        float val;
        if (n == ment) {
            val = docp[(long long)b * cD + d];
        } else if (d < cH) {
            float acc = 0.f;
            for (int s = 0; s < cS; ++s) {
                int p = pos[s];
                acc += tokf[((long long)(b * cT + p)) * cH + d] * sm[s];
            }
            val = (acc * inv + doccls[(long long)b * cH + d]) * nm;
        } else {
            val = nerEmb[(long long)ner * cNERD + (d - cH)] * nm;
        }
        out[d] = val;
    }
}

// X (B,N,RS,D): zero slots r<15, slot 15 = h
__global__ void k_init_X(const float* __restrict__ h, float* __restrict__ X, long long n) {
    long long i = (long long)blockIdx.x * blockDim.x + threadIdx.x;
    long long stride = (long long)gridDim.x * blockDim.x;
    for (; i < n; i += stride) {
        int d = (int)(i % cD);
        long long t = i / cD;
        int r  = (int)(t % cRS); t /= cRS;
        int nn = (int)(t % cN);
        int b  = (int)(t / cN);
        X[i] = (r == cR) ? h[((long long)(b * cN + nn)) * cD + d] : 0.f;
    }
}

// scatter: X[b, dst, rel, :] += h[b, src, :]
__global__ void k_scatter(const float* __restrict__ h, const int* __restrict__ esrc,
                          const int* __restrict__ edst, const int* __restrict__ erel,
                          float* __restrict__ X) {
    int blk = blockIdx.x;
    int b = blk / cE, e = blk % cE;
    int src = esrc[b * cE + e];
    int dst = edst[b * cE + e];
    int rel = erel[b * cE + e];
    const float* hs = h + ((long long)(b * cN + src)) * cD;
    float* xd = X + (((long long)(b * cN + dst)) * cRS + rel) * cD;
    for (int d = threadIdx.x; d < cD; d += blockDim.x)
        atomicAdd(&xd[d], hs[d]);
}

// ent[b,e,:] = masked mean over mentions of concat(h1,h2) with zero row at idx 0
__global__ void k_entity_agg(const float* __restrict__ h1, const float* __restrict__ h2,
                             const int* __restrict__ e2m, const float* __restrict__ e2mm,
                             float* __restrict__ ent) {
    int blk = blockIdx.x;
    int b = blk / cNE, e = blk % cNE;
    const int*   idx = e2m  + ((long long)(b * cNE + e)) * cMM;
    const float* msk = e2mm + ((long long)(b * cNE + e)) * cMM;
    float msum = 0.f;
    for (int m = 0; m < cMM; ++m) msum += msk[m];
    float inv = 1.f / (msum + 1e-7f);
    float* out = ent + ((long long)(b * cNE + e)) * cEE;
    for (int d = threadIdx.x; d < cEE; d += blockDim.x) {
        float acc = 0.f;
        for (int m = 0; m < cMM; ++m) {
            int id = idx[m];
            float v = 0.f;
            if (id > 0) {
                long long row = (long long)(b * cN + (id - 1));
                v = (d < cD) ? h1[row * cD + d] : h2[row * cD + (d - cD)];
            }
            acc += msk[m] * v;
        }
        out[d] = acc * inv;
    }
}

// feat rows: [hf, tf, |hf-tf|, hf*tf]  -> bf16 (B*P, 6400)
__global__ void k_build_feat(const float* __restrict__ ent, const int* __restrict__ pairs,
                             bf16_t* __restrict__ feat) {
    int blk = blockIdx.x;
    int b = blk / cP, p = blk % cP;
    int i0 = pairs[((long long)(b * cP + p)) * 2 + 0];
    int i1 = pairs[((long long)(b * cP + p)) * 2 + 1];
    const float* hf = ent + ((long long)(b * cNE + i0)) * cEE;
    const float* tf = ent + ((long long)(b * cNE + i1)) * cEE;
    bf16_t* row = feat + ((long long)(b * cP + p)) * cF4;
    for (int d = threadIdx.x; d < cEE; d += blockDim.x) {
        float a = hf[d], c = tf[d];
        row[d]            = (bf16_t)a;
        row[cEE + d]      = (bf16_t)c;
        row[2 * cEE + d]  = (bf16_t)fabsf(a - c);
        row[3 * cEE + d]  = (bf16_t)(a * c);
    }
}

// ---------------- tiled bf16 WMMA GEMM ----------------
// C[M,N] = A[M,K] @ Bt[N,K]^T (+bias, opt relu), f32 accum, f32 or bf16 out.
// Bt is the transposed (N-major) weight matrix -> both operands stream as b128.
// 256 threads = 8 waves (4 along M x 2 along N).
// Async path: double-buffered LDS fed by GLOBAL_LOAD_ASYNC_TO_LDS_B128, waited
// with S_WAIT_ASYNCCNT (own tile done, next tile in flight).
// Fallback: register-staged software pipeline via regular b128 loads.
template<int BM, int BN>
__launch_bounds__(256)
__global__ void k_gemm(const bf16_t* __restrict__ A, long long lda, long long strideA,
                       const bf16_t* __restrict__ Bt, long long ldb,
                       void* __restrict__ C, long long ldc, long long strideC,
                       const float* __restrict__ bias,
                       int M, int N, int K, int relu, int out_bf16) {
    constexpr int WM = 4, WN = 2;
    constexpr int WROWS = BM / WM;
    constexpr int WCOLS = BN / WN;
    constexpr int MT = WROWS / 16;
    constexpr int NT = WCOLS / 16;
    constexpr int AREG = BM * 4 / 256;    // 16B chunks per thread per tile
    constexpr int BREG = BN * 4 / 256;

    const int tid  = threadIdx.x;
    const int lane = tid & 31;
    const int wave = tid >> 5;
    const int waveM = wave % WM;
    const int waveN = wave / WM;
    const int hi   = lane >> 4;
    const int lrow = lane & 15;

    const int m0 = blockIdx.y * BM;
    const int n0 = blockIdx.x * BN;
    const long long aOff = (long long)blockIdx.z * strideA;
    const long long cOff = (long long)blockIdx.z * strideC;

    // per-thread staging coordinates
    int am[AREG], akc[AREG], bn[BREG], bkc[BREG];
    for (int j = 0; j < AREG; ++j) { int i = tid + j * 256; am[j] = i >> 2; akc[j] = i & 3; }
    for (int j = 0; j < BREG; ++j) { int i = tid + j * 256; bn[j] = i >> 2; bkc[j] = i & 3; }

    v8f acc[MT][NT];
    for (int mt = 0; mt < MT; ++mt)
        for (int nt = 0; nt < NT; ++nt)
            for (int q = 0; q < 8; ++q) acc[mt][nt][q] = 0.f;

#if USE_ASYNC_LDS
    constexpr int LDSK = 40;              // 16B-aligned chunk slots
    __shared__ bf16_t Alds[2][BM * LDSK];
    __shared__ bf16_t Blds[2][BN * LDSK];

    // clamp OOB B rows (cols >= N are never stored, garbage is harmless) so
    // every wave issues the same number of async ops -> uniform ASYNCcnt.
    int bnc[BREG];
    for (int j = 0; j < BREG; ++j) {
        int n = n0 + bn[j];
        bnc[j] = (n < N) ? n : (N - 1);
    }
    auto issue = [&](int buf, int k0) {
        for (int j = 0; j < AREG; ++j)
            __builtin_amdgcn_global_load_async_to_lds_b128(
                to_glb(A + aOff + (long long)(m0 + am[j]) * lda + k0 + akc[j] * 8),
                to_lds(&Alds[buf][am[j] * LDSK + akc[j] * 8]), 0, 0);
        for (int j = 0; j < BREG; ++j)
            __builtin_amdgcn_global_load_async_to_lds_b128(
                to_glb(Bt + (long long)bnc[j] * ldb + k0 + bkc[j] * 8),
                to_lds(&Blds[buf][bn[j] * LDSK + bkc[j] * 8]), 0, 0);
    };

    issue(0, 0);
    int cur = 0;
    for (int k0 = 0; k0 < K; k0 += 32) {
        __syncthreads();                 // readers of buf cur^1 (previous tile) done
        if (k0 + 32 < K) {
            issue(cur ^ 1, k0 + 32);
            __builtin_amdgcn_s_wait_asynccnt(AREG + BREG);  // own tile landed
        } else {
            __builtin_amdgcn_s_wait_asynccnt(0);
        }
        __syncthreads();                 // whole tile visible to all waves

        v16bf afr[MT], bfr[NT];
        for (int mt = 0; mt < MT; ++mt) {
            const bf16_t* p = &Alds[cur][(waveM * WROWS + mt * 16 + lrow) * LDSK];
            union { v16bf v; uint2 u[4]; } ua;
            ua.u[0] = *(const uint2*)(p + 8 * hi);
            ua.u[1] = *(const uint2*)(p + 8 * hi + 4);
            ua.u[2] = *(const uint2*)(p + 16 + 8 * hi);
            ua.u[3] = *(const uint2*)(p + 16 + 8 * hi + 4);
            afr[mt] = ua.v;
        }
        for (int nt = 0; nt < NT; ++nt) {
            const bf16_t* p = &Blds[cur][(waveN * WCOLS + nt * 16 + lrow) * LDSK];
            union { v16bf v; uint2 u[4]; } ub;
            ub.u[0] = *(const uint2*)(p + 8 * hi);
            ub.u[1] = *(const uint2*)(p + 8 * hi + 4);
            ub.u[2] = *(const uint2*)(p + 16 + 8 * hi);
            ub.u[3] = *(const uint2*)(p + 16 + 8 * hi + 4);
            bfr[nt] = ub.v;
        }
        for (int mt = 0; mt < MT; ++mt)
            for (int nt = 0; nt < NT; ++nt)
                acc[mt][nt] = __builtin_amdgcn_wmma_f32_16x16x32_bf16(
                    false, afr[mt], false, bfr[nt], (short)0, acc[mt][nt], false, false);
        cur ^= 1;
    }
#else
    constexpr int LDSK = 36;
    __shared__ bf16_t Alds[BM * LDSK];
    __shared__ bf16_t Blds[BN * LDSK];

    uint4 ra[AREG], rb[BREG];
    auto fetch = [&](int k0) {
        for (int j = 0; j < AREG; ++j)
            ra[j] = *(const uint4*)(A + aOff + (long long)(m0 + am[j]) * lda + k0 + akc[j] * 8);
        for (int j = 0; j < BREG; ++j) {
            if (n0 + bn[j] < N)
                rb[j] = *(const uint4*)(Bt + (long long)(n0 + bn[j]) * ldb + k0 + bkc[j] * 8);
            else
                rb[j] = uint4{0u, 0u, 0u, 0u};
        }
    };

    fetch(0);
    for (int k0 = 0; k0 < K; k0 += 32) {
        for (int j = 0; j < AREG; ++j) {
            union { uint4 u4; uint2 u2[2]; } v; v.u4 = ra[j];
            bf16_t* dst = &Alds[am[j] * LDSK + akc[j] * 8];
            *(uint2*)(dst)     = v.u2[0];
            *(uint2*)(dst + 4) = v.u2[1];
        }
        for (int j = 0; j < BREG; ++j) {
            union { uint4 u4; uint2 u2[2]; } v; v.u4 = rb[j];
            bf16_t* dst = &Blds[bn[j] * LDSK + bkc[j] * 8];
            *(uint2*)(dst)     = v.u2[0];
            *(uint2*)(dst + 4) = v.u2[1];
        }
        __syncthreads();
        if (k0 + 32 < K) fetch(k0 + 32);

        v16bf afr[MT], bfr[NT];
        for (int mt = 0; mt < MT; ++mt) {
            const bf16_t* p = &Alds[(waveM * WROWS + mt * 16 + lrow) * LDSK];
            union { v16bf v; uint2 u[4]; } ua;
            ua.u[0] = *(const uint2*)(p + 8 * hi);
            ua.u[1] = *(const uint2*)(p + 8 * hi + 4);
            ua.u[2] = *(const uint2*)(p + 16 + 8 * hi);
            ua.u[3] = *(const uint2*)(p + 16 + 8 * hi + 4);
            afr[mt] = ua.v;
        }
        for (int nt = 0; nt < NT; ++nt) {
            const bf16_t* p = &Blds[(waveN * WCOLS + nt * 16 + lrow) * LDSK];
            union { v16bf v; uint2 u[4]; } ub;
            ub.u[0] = *(const uint2*)(p + 8 * hi);
            ub.u[1] = *(const uint2*)(p + 8 * hi + 4);
            ub.u[2] = *(const uint2*)(p + 16 + 8 * hi);
            ub.u[3] = *(const uint2*)(p + 16 + 8 * hi + 4);
            bfr[nt] = ub.v;
        }
        for (int mt = 0; mt < MT; ++mt)
            for (int nt = 0; nt < NT; ++nt)
                acc[mt][nt] = __builtin_amdgcn_wmma_f32_16x16x32_bf16(
                    false, afr[mt], false, bfr[nt], (short)0, acc[mt][nt], false, false);
        __syncthreads();
    }
#endif

    // epilogue: C layout (VGPR j, lane l) -> M = j + 8*hi, N = lane&15
    for (int mt = 0; mt < MT; ++mt) {
        for (int nt = 0; nt < NT; ++nt) {
            int col = n0 + waveN * WCOLS + nt * 16 + lrow;
            if (col >= N) continue;
            int rowBase = m0 + waveM * WROWS + mt * 16 + 8 * hi;
            float bv = bias ? bias[col] : 0.f;
            for (int j = 0; j < 8; ++j) {
                float v = acc[mt][nt][j] + bv;
                if (relu) v = v > 0.f ? v : 0.f;
                long long off = cOff + (long long)(rowBase + j) * ldc + col;
                if (out_bf16) ((bf16_t*)C)[off] = (bf16_t)v;
                else          ((float*)C)[off]  = v;
            }
        }
    }
}

// ---------------- host orchestration ----------------
extern "C" void kernel_launch(void* const* d_in, const int* in_sizes, int n_in,
                              void* d_out, int out_size, void* d_ws, size_t ws_size,
                              hipStream_t stream) {
    const float* tokf    = (const float*)d_in[0];
    const float* doccls  = (const float*)d_in[1];
    const int*   spanpos = (const int*)  d_in[2];
    const float* spanmsk = (const float*)d_in[3];
    const int*   nner    = (const int*)  d_in[4];
    const float* nmask   = (const float*)d_in[5];
    const int*   e2m     = (const int*)  d_in[6];
    const float* e2mm    = (const float*)d_in[7];
    const int*   pairs   = (const int*)  d_in[8];
    const int*   esrc    = (const int*)  d_in[9];
    const int*   edst    = (const int*)  d_in[10];
    const int*   erel    = (const int*)  d_in[11];
    const int*   mentnum = (const int*)  d_in[12];
    const float* nerEmb  = (const float*)d_in[13];
    const float* docW    = (const float*)d_in[14];
    const float* docb    = (const float*)d_in[15];
    const float* Wrel    = (const float*)d_in[16];
    const float* Wself   = (const float*)d_in[17];
    const float* gnnb    = (const float*)d_in[18];
    const float* W1      = (const float*)d_in[19];
    const float* b1      = (const float*)d_in[20];
    const float* W2      = (const float*)d_in[21];
    const float* b2      = (const float*)d_in[22];

    char* ws = (char*)d_ws;
    size_t off = 0;
    auto alloc = [&](size_t bytes) -> char* {
        char* p = ws + off;
        off = (off + bytes + 255) & ~(size_t)255;
        return p;
    };
    const long long nWcat = (long long)cL * cRS * cD * cD;   // 20,480,000
    const long long nW1   = (long long)cF4 * cHID;           // 20,480,000
    const long long nW2   = (long long)cHID * cOUT;          // 310,400
    const long long nH    = (long long)cB * cN * cD;         // 819,200
    const long long nX    = (long long)cB * cN * cRS * cD;   // 13,107,200

    bf16_t* wcatT = (bf16_t*)alloc(nWcat * 2);   // (L, 800, 12800)
    bf16_t* w1T   = (bf16_t*)alloc(nW1 * 2);     // (3200, 6400)
    bf16_t* w2T   = (bf16_t*)alloc(nW2 * 2);     // (97, 3200)
    float*  docp  = (float*) alloc((long long)cB * cD * 4);
    float*  h0    = (float*) alloc(nH * 4);
    float*  h1    = (float*) alloc(nH * 4);
    float*  h2    = (float*) alloc(nH * 4);
    float*  Xf    = (float*) alloc(nX * 4);      // reused later for feat (bf16, same bytes)
    bf16_t* Xbf   = (bf16_t*)alloc(nX * 2);      // reused later for hidden (bf16, same bytes)
    float*  ent   = (float*) alloc((long long)cB * cNE * cEE * 4);
    bf16_t* featbf = (bf16_t*)Xf;
    bf16_t* hidbf  = (bf16_t*)Xbf;

    // weights -> bf16, pre-transposed to N-major for b128 GEMM staging
    {
        dim3 g(cD / 32, (int)(cKGNN / 32), cL);
        k_build_wcatT<<<g, 256, 0, stream>>>(Wrel, Wself, wcatT);
    }
    {
        dim3 g(cHID / 32, cF4 / 32, 1);
        k_transpose_bf16<<<g, 256, 0, stream>>>(W1, w1T, cF4, cHID);
    }
    {
        dim3 g((cOUT + 31) / 32, cHID / 32, 1);
        k_transpose_bf16<<<g, 256, 0, stream>>>(W2, w2T, cHID, cOUT);
    }

    // node features
    k_doc_proj   <<<(cB * cD + 255) / 256, 256, 0, stream>>>(doccls, docW, docb, docp);
    k_build_nodes<<<cB * cN, 256, 0, stream>>>(tokf, doccls, spanpos, spanmsk, nner, nmask,
                                               mentnum, nerEmb, docp, h0);

    // GNN layers: pre-aggregate then one fat GEMM per sample
    for (int l = 0; l < cL; ++l) {
        const float* hin  = (l == 0) ? h0 : h1;
        float*       hout = (l == 0) ? h1 : h2;
        k_init_X <<<8192, 256, 0, stream>>>(hin, Xf, nX);
        k_scatter<<<cB * cE, 256, 0, stream>>>(hin, esrc, edst, erel, Xf);
        k_f32_to_bf16<<<8192, 256, 0, stream>>>(Xf, Xbf, nX);
        dim3 g((cD + 63) / 64, cN / 64, cB);
        k_gemm<64, 64><<<g, 256, 0, stream>>>(
            Xbf, cKGNN, (long long)cN * cKGNN,
            wcatT + (long long)l * cD * cKGNN, cKGNN,
            hout, cD, (long long)cN * cD,
            gnnb + (long long)l * cD, cN, cD, (int)cKGNN, /*relu=*/1, /*bf16=*/0);
    }

    // entity pooling + pair features
    k_entity_agg<<<cB * cNE, 256, 0, stream>>>(h1, h2, e2m, e2mm, ent);
    k_build_feat<<<cB * cP, 256, 0, stream>>>(ent, pairs, featbf);

    // MLP1: (4096 x 6400) @ (6400 x 3200) + b1, relu -> bf16 hidden
    {
        dim3 g(cHID / 128, (cB * cP) / 128, 1);
        k_gemm<128, 128><<<g, 256, 0, stream>>>(
            featbf, cF4, 0, w1T, cF4,
            hidbf, cHID, 0, b1, cB * cP, cHID, cF4, /*relu=*/1, /*bf16=*/1);
    }
    // MLP2: (4096 x 3200) @ (3200 x 97) + b2 -> f32 d_out
    {
        dim3 g((cOUT + 63) / 64, (cB * cP) / 64, 1);
        k_gemm<64, 64><<<g, 256, 0, stream>>>(
            hidbf, cHID, 0, w2T, cHID,
            d_out, cOUT, 0, b2, cB * cP, cOUT, cHID, /*relu=*/0, /*bf16=*/0);
    }
}
